// BlockMamba_10514079941290
// MI455X (gfx1250) — compile-verified, matrix operand order checked by reading, not run on hardware
//
#include <hip/hip_runtime.h>
#include <hip/hip_bf16.h>
#include <math.h>

// ---------------- problem constants ----------------
#define Bsz 4
#define Nseq 2048
#define Dm 384
#define DIN 768
#define DSTATE 16
#define DCONV 4
#define DTRANK 24
#define Kn 5
#define HID 384
#define EPSV 1e-5f
#define ROWS (Bsz * Nseq) // 8192

// ---------------- WMMA types ----------------
typedef __attribute__((ext_vector_type(16))) __bf16 v16bf;
typedef __attribute__((ext_vector_type(8))) float v8f;

struct U4 { unsigned int a, b, c, d; };
union Frag { v16bf v; U4 q[2]; };

__device__ __forceinline__ unsigned short f2bf(float f) {
  unsigned int u = __builtin_bit_cast(unsigned int, f);
  u += 0x7FFFu + ((u >> 16) & 1u);  // round-to-nearest-even
  return (unsigned short)(u >> 16);
}
__device__ __forceinline__ float siluf(float x) { return x / (1.0f + __expf(-x)); }
__device__ __forceinline__ float geluf(float x) { return 0.5f * x * (1.0f + erff(x * 0.70710678118654752f)); }
__device__ __forceinline__ float softplusf(float x) { return (x > 20.0f) ? x : log1pf(__expf(x)); }

// =====================================================================
// bf16-WMMA GEMM, pre-converted operands:
//   C[M x N](fp32) = A[M x K](bf16) * W[N x K](bf16)^T (+bias +res)
// block = 256 threads (8 waves), block tile 128x128, wave tile 32x64,
// K step 32, ping-pong double-buffered LDS, 1 barrier per K step.
// Epilogue specialised at compile time (no per-element branches,
// no dead +0 adds in the no-bias variants).
// =====================================================================
template <bool HAS_BIAS, bool HAS_RES>
__global__ __launch_bounds__(256) void gemm_bf16_wmma(
    const unsigned short* __restrict__ A, const unsigned short* __restrict__ W,
    float* __restrict__ C, const float* __restrict__ bias,
    const float* __restrict__ res, int M, int N, int Kd) {
  __shared__ __align__(16) unsigned short sA[2][128 * 40];
  __shared__ __align__(16) unsigned short sB[2][128 * 40];

  const int tid  = threadIdx.x;
  const int nblk = blockIdx.x;
  const int mblk = blockIdx.y;
  const int lane = tid & 31;
  const int wv   = tid >> 5;    // wave 0..7
  const int wm   = wv >> 1;     // 0..3  (M direction, 32 rows each)
  const int wn   = wv & 1;      // 0..1  (N direction, 64 cols each)
  const int r16  = lane & 15;
  const int hi   = lane >> 4;
  const int kb   = hi * 8;      // K base inside fragment (halfs)

  // staging: thread owns (row rs, 16-half chunk hs) of both tiles
  const int rs = tid >> 1;            // 0..127
  const int hs = (tid & 1) * 16;      // 0 or 16
  const unsigned short* gA = A + (size_t)(mblk * 128 + rs) * Kd + hs;
  const unsigned short* gB = W + (size_t)(nblk * 128 + rs) * Kd + hs;
  const int soff = rs * 40 + hs;

  // prologue: stage K-step 0 into buffer 0
  {
    U4 a0 = *(const U4*)gA, a1 = *(const U4*)(gA + 8);
    U4 b0 = *(const U4*)gB, b1 = *(const U4*)(gB + 8);
    *(U4*)&sA[0][soff] = a0; *(U4*)&sA[0][soff + 8] = a1;
    *(U4*)&sB[0][soff] = b0; *(U4*)&sB[0][soff + 8] = b1;
  }
  __syncthreads();

  v8f acc[2][4];
#pragma unroll
  for (int mt = 0; mt < 2; ++mt)
#pragma unroll
    for (int nt = 0; nt < 4; ++nt) acc[mt][nt] = v8f{};

  const int nsteps = Kd >> 5;
  for (int ks = 0; ks < nsteps; ++ks) {
    const int cur = ks & 1;
    const bool more = (ks + 1) < nsteps;
    // issue global loads for next K step early (hidden under WMMAs)
    U4 a0, a1, b0, b1;
    if (more) {
      const unsigned short* pa = gA + (size_t)(ks + 1) * 32;
      const unsigned short* pb = gB + (size_t)(ks + 1) * 32;
      a0 = *(const U4*)pa; a1 = *(const U4*)(pa + 8);
      b0 = *(const U4*)pb; b1 = *(const U4*)(pb + 8);
    }
    // fragments from current buffer
    Frag af[2], bfr[4];
#pragma unroll
    for (int mt = 0; mt < 2; ++mt) {
      const unsigned short* p = &sA[cur][(wm * 32 + mt * 16 + r16) * 40 + kb];
      af[mt].q[0] = *(const U4*)p;
      af[mt].q[1] = *(const U4*)(p + 16);
    }
#pragma unroll
    for (int nt = 0; nt < 4; ++nt) {
      const unsigned short* p = &sB[cur][(wn * 64 + nt * 16 + r16) * 40 + kb];
      bfr[nt].q[0] = *(const U4*)p;
      bfr[nt].q[1] = *(const U4*)(p + 16);
    }
    // 8 WMMAs per wave per K step
#pragma unroll
    for (int mt = 0; mt < 2; ++mt)
#pragma unroll
      for (int nt = 0; nt < 4; ++nt)
        acc[mt][nt] = __builtin_amdgcn_wmma_f32_16x16x32_bf16(
            false, af[mt].v, false, bfr[nt].v, (short)0, acc[mt][nt], false, false);
    // park next tile into the other buffer
    if (more) {
      const int nxt = cur ^ 1;
      *(U4*)&sA[nxt][soff] = a0; *(U4*)&sA[nxt][soff + 8] = a1;
      *(U4*)&sB[nxt][soff] = b0; *(U4*)&sB[nxt][soff + 8] = b1;
    }
    __syncthreads();
  }

  // epilogue: C/D layout -> VGPR r holds M = r + hi*8, N = r16
#pragma unroll
  for (int nt = 0; nt < 4; ++nt) {
    const int gn = nblk * 128 + wn * 64 + nt * 16 + r16;
    float bv = 0.0f;
    if (HAS_BIAS) bv = bias[gn];     // invariant over the 8 M rows
#pragma unroll
    for (int mt = 0; mt < 2; ++mt) {
      const int gm0 = mblk * 128 + wm * 32 + mt * 16 + hi * 8;
      float rv[8];
      if (HAS_RES) {
#pragma unroll
        for (int r = 0; r < 8; ++r) rv[r] = res[(size_t)(gm0 + r) * N + gn];
      }
#pragma unroll
      for (int r = 0; r < 8; ++r) {
        float v = acc[mt][nt][r];
        if (HAS_BIAS) v += bv;
        if (HAS_RES)  v += rv[r];
        C[(size_t)(gm0 + r) * N + gn] = v;
      }
    }
  }
}

// =====================================================================
// LayerNorm over D=384 -> bf16 output; one row per block, 128 threads
// =====================================================================
__global__ __launch_bounds__(128) void ln_kernel_bf16(
    const float* __restrict__ x, const float* __restrict__ g,
    const float* __restrict__ b, unsigned short* __restrict__ out) {
  __shared__ float red[8];
  const int row = blockIdx.x, tid = threadIdx.x;
  float v[3], s = 0.f, ss = 0.f;
#pragma unroll
  for (int j = 0; j < 3; ++j) {
    v[j] = x[(size_t)row * Dm + tid + j * 128];
    s += v[j]; ss += v[j] * v[j];
  }
#pragma unroll
  for (int off = 16; off; off >>= 1) {
    s  += __shfl_xor(s, off, 32);
    ss += __shfl_xor(ss, off, 32);
  }
  const int wid = tid >> 5;
  if ((tid & 31) == 0) { red[wid] = s; red[4 + wid] = ss; }
  __syncthreads();
  s  = red[0] + red[1] + red[2] + red[3];
  ss = red[4] + red[5] + red[6] + red[7];
  const float mu = s * (1.0f / Dm);
  const float var = ss * (1.0f / Dm) - mu * mu;
  const float rs = rsqrtf(var + EPSV);
#pragma unroll
  for (int j = 0; j < 3; ++j) {
    const int e = tid + j * 128;
    out[(size_t)row * Dm + e] = f2bf((v[j] - mu) * rs * g[e] + b[e]);
  }
}

// =====================================================================
// weight prep: fp32 -> bf16 (contiguous)
// =====================================================================
__global__ __launch_bounds__(256) void f32_to_bf16_kernel(
    const float* __restrict__ src, unsigned short* __restrict__ dst, int n) {
  const int i = blockIdx.x * 256 + threadIdx.x;
  if (i < n) dst[i] = f2bf(src[i]);
}

// fc1 split: W1a = fc1_w[:, :384] ; Wd = fc1_w[:, 384:] - fc1_w[:, :384]
__global__ __launch_bounds__(256) void fc1_prep_kernel(
    const float* __restrict__ fc1w, unsigned short* __restrict__ w1a,
    unsigned short* __restrict__ wd) {
  const int i = blockIdx.x * 256 + threadIdx.x;
  if (i >= HID * Dm) return;
  const int h = i / Dm, c = i % Dm;
  const float a = fc1w[(size_t)h * (2 * Dm) + c];
  const float b = fc1w[(size_t)h * (2 * Dm) + Dm + c];
  w1a[i] = f2bf(a);
  wd[i]  = f2bf(b - a);
}

// =====================================================================
// Depthwise causal conv (DCONV=4) + SiLU ; xi = xz[..., 0:768]
// =====================================================================
__global__ __launch_bounds__(256) void conv_silu_kernel(
    const float* __restrict__ xz, const float* __restrict__ cw,
    const float* __restrict__ cb, float* __restrict__ xc) {
  const size_t gid = (size_t)blockIdx.x * 256 + threadIdx.x;
  if (gid >= (size_t)ROWS * DIN) return;
  const int d = (int)(gid % DIN);
  const int row = (int)(gid / DIN);
  const int t = row & (Nseq - 1);
  float acc = cb[d];
#pragma unroll
  for (int j = 0; j < DCONV; ++j) {
    const int tt = t - (DCONV - 1) + j;
    if (tt >= 0) acc += cw[d * DCONV + j] * xz[(size_t)(row - (t - tt)) * (2 * DIN) + d];
  }
  xc[gid] = siluf(acc);
}

// =====================================================================
// x_dbl = xc @ x_proj_w^T  (K=768, N=56) : one row per 64-thread block
// =====================================================================
__global__ __launch_bounds__(64) void xdbl_kernel(
    const float* __restrict__ xc, const float* __restrict__ w,
    float* __restrict__ xdbl) {
  __shared__ float srow[DIN];
  const int row = blockIdx.x, tid = threadIdx.x;
  for (int i = tid; i < DIN; i += 64) srow[i] = xc[(size_t)row * DIN + i];
  __syncthreads();
  if (tid < DTRANK + 2 * DSTATE) {
    const float* wr = w + (size_t)tid * DIN;
    float acc = 0.f;
#pragma unroll 8
    for (int k = 0; k < DIN; ++k) acc += srow[k] * wr[k];
    xdbl[(size_t)row * 56 + tid] = acc;
  }
}

// =====================================================================
// dt = softplus(dt_r @ dt_proj_w^T + dt_proj_b)  (K=24, N=768)
// =====================================================================
__global__ __launch_bounds__(256) void dt_kernel(
    const float* __restrict__ xdbl, const float* __restrict__ w,
    const float* __restrict__ bias, float* __restrict__ dt) {
  __shared__ float sr[DTRANK];
  const int row = blockIdx.x, tid = threadIdx.x;
  if (tid < DTRANK) sr[tid] = xdbl[(size_t)row * 56 + tid];
  __syncthreads();
  for (int d = tid; d < DIN; d += 256) {
    float acc = bias[d];
#pragma unroll
    for (int r = 0; r < DTRANK; ++r) acc += sr[r] * w[d * DTRANK + r];
    dt[(size_t)row * DIN + d] = softplusf(acc);
  }
}

// =====================================================================
// Selective scan over N=2048 + Dskip + silu(z) gate -> bf16 yact
// One thread per (b,d) channel, 16 states in registers; B/C vectors
// staged in LDS in 64-step chunks.
// =====================================================================
#define TCHUNK 64
__global__ __launch_bounds__(256) void scan_kernel(
    const float* __restrict__ dt, const float* __restrict__ xdbl,
    const float* __restrict__ xc, const float* __restrict__ xz,
    const float* __restrict__ A_log, const float* __restrict__ Dskip,
    unsigned short* __restrict__ yact) {
  __shared__ float sBC[TCHUNK * 32];
  const int blk = blockIdx.x;            // 12 blocks: 3 per batch
  const int b = blk / 3;
  const int d = (blk % 3) * 256 + threadIdx.x;
  const int tid = threadIdx.x;
  const size_t browS = (size_t)b * Nseq;

  float As[DSTATE], h[DSTATE];
#pragma unroll
  for (int s = 0; s < DSTATE; ++s) { As[s] = -__expf(A_log[d * DSTATE + s]); h[s] = 0.f; }
  const float dsk = Dskip[d];

  for (int t0 = 0; t0 < Nseq; t0 += TCHUNK) {
    for (int i = tid; i < TCHUNK * 32; i += 256) {
      const int ts = i >> 5, s = i & 31;
      sBC[i] = xdbl[(browS + t0 + ts) * 56 + DTRANK + s];
    }
    __syncthreads();
    for (int ts = 0; ts < TCHUNK; ++ts) {
      const size_t rr = browS + t0 + ts;
      const float dt_v = dt[rr * DIN + d];
      const float u    = xc[rr * DIN + d];
      const float du   = dt_v * u;
      const float* bc  = &sBC[ts * 32];
      float y = 0.f;
#pragma unroll
      for (int s = 0; s < DSTATE; ++s) {
        const float dA = __expf(dt_v * As[s]);
        h[s] = dA * h[s] + du * bc[s];
        y += h[s] * bc[16 + s];
      }
      const float z = xz[rr * (2 * DIN) + DIN + d];
      yact[rr * DIN + d] = f2bf((y + u * dsk) * siluf(z));
    }
    __syncthreads();
  }
}

// =====================================================================
// u[row][h] = max_k gelu(P[idx_k][h] + Q[row][h] + fc1_b[h]) -> bf16
// =====================================================================
__global__ __launch_bounds__(128) void lcffn_gather_kernel(
    const float* __restrict__ P, const float* __restrict__ Q,
    const float* __restrict__ fc1b, const int* __restrict__ idx,
    unsigned short* __restrict__ u) {
  const int row = blockIdx.x, tid = threadIdx.x;
  const int b = row >> 11;  // row / 2048
  int iv[Kn];
#pragma unroll
  for (int k = 0; k < Kn; ++k) iv[k] = idx[(size_t)row * Kn + k];
#pragma unroll
  for (int j = 0; j < 3; ++j) {
    const int hcol = tid + j * 128;
    const float q = Q[(size_t)row * HID + hcol] + fc1b[hcol];
    float m = -INFINITY;
#pragma unroll
    for (int k = 0; k < Kn; ++k) {
      const float v = geluf(P[((size_t)b * Nseq + iv[k]) * HID + hcol] + q);
      m = fmaxf(m, v);
    }
    u[(size_t)row * HID + hcol] = f2bf(m);
  }
}

// =====================================================================
// host launcher
// =====================================================================
extern "C" void kernel_launch(void* const* d_in, const int* in_sizes, int n_in,
                              void* d_out, int out_size, void* d_ws, size_t ws_size,
                              hipStream_t stream) {
  const float* x        = (const float*)d_in[0];
  const int*   idx      = (const int*)d_in[1];
  const float* ln1_g    = (const float*)d_in[2];
  const float* ln1_b    = (const float*)d_in[3];
  const float* ln2_g    = (const float*)d_in[4];
  const float* ln2_b    = (const float*)d_in[5];
  const float* in_proj  = (const float*)d_in[6];
  const float* conv_w   = (const float*)d_in[7];
  const float* conv_b   = (const float*)d_in[8];
  const float* x_proj   = (const float*)d_in[9];
  const float* dtp_w    = (const float*)d_in[10];
  const float* dtp_b    = (const float*)d_in[11];
  const float* A_log    = (const float*)d_in[12];
  const float* Dskip    = (const float*)d_in[13];
  const float* out_proj = (const float*)d_in[14];
  const float* fc1_w    = (const float*)d_in[15];
  const float* fc1_b    = (const float*)d_in[16];
  const float* fc2_w    = (const float*)d_in[17];
  const float* fc2_b    = (const float*)d_in[18];
  float* out = (float*)d_out;

  char* wsb = (char*)d_ws;
  size_t off = 0;
  auto alloc = [&](size_t bytes) -> void* {
    off = (off + 255) & ~(size_t)255;
    void* p = wsb + off; off += bytes; return p;
  };
  typedef unsigned short bf;
  bf*    x1b   = (bf*)alloc((size_t)ROWS * Dm * 2);
  bf*    inpb  = (bf*)alloc((size_t)2 * DIN * Dm * 2);
  float* xz    = (float*)alloc((size_t)ROWS * 2 * DIN * 4);
  float* xc    = (float*)alloc((size_t)ROWS * DIN * 4);
  float* xdbl  = (float*)alloc((size_t)ROWS * 56 * 4);
  float* dtb   = (float*)alloc((size_t)ROWS * DIN * 4);
  bf*    yactb = (bf*)alloc((size_t)ROWS * DIN * 2);
  bf*    outpb = (bf*)alloc((size_t)Dm * DIN * 2);
  float* x2    = (float*)alloc((size_t)ROWS * Dm * 4);
  bf*    x3b   = (bf*)alloc((size_t)ROWS * Dm * 2);
  bf*    w1ab  = (bf*)alloc((size_t)HID * Dm * 2);
  bf*    wdb   = (bf*)alloc((size_t)HID * Dm * 2);
  float* Pm    = (float*)alloc((size_t)ROWS * HID * 4);
  float* Qm    = (float*)alloc((size_t)ROWS * HID * 4);
  bf*    umb   = (bf*)alloc((size_t)ROWS * HID * 2);
  bf*    fc2b  = (bf*)alloc((size_t)Dm * HID * 2);

  // weight prep (small, runs before LN1 on the same stream)
  f32_to_bf16_kernel<<<(2 * DIN * Dm + 255) / 256, 256, 0, stream>>>(in_proj, inpb, 2 * DIN * Dm);
  f32_to_bf16_kernel<<<(Dm * DIN + 255) / 256, 256, 0, stream>>>(out_proj, outpb, Dm * DIN);
  f32_to_bf16_kernel<<<(Dm * HID + 255) / 256, 256, 0, stream>>>(fc2_w, fc2b, Dm * HID);
  fc1_prep_kernel<<<(HID * Dm + 255) / 256, 256, 0, stream>>>(fc1_w, w1ab, wdb);

  // 1) x1 = LN1(x) -> bf16
  ln_kernel_bf16<<<ROWS, 128, 0, stream>>>(x, ln1_g, ln1_b, x1b);
  // 2) xz = x1 @ in_proj^T   (8192x1536, K=384)
  gemm_bf16_wmma<false, false><<<dim3((2 * DIN) / 128, ROWS / 128), 256, 0, stream>>>(
      x1b, inpb, xz, nullptr, nullptr, ROWS, 2 * DIN, Dm);
  // 3) depthwise conv + silu
  conv_silu_kernel<<<(ROWS * DIN + 255) / 256, 256, 0, stream>>>(xz, conv_w, conv_b, xc);
  // 4) x_dbl = xc @ x_proj^T (N=56)
  xdbl_kernel<<<ROWS, 64, 0, stream>>>(xc, x_proj, xdbl);
  // 5) dt = softplus(dt_r @ dt_proj^T + b)
  dt_kernel<<<ROWS, 256, 0, stream>>>(xdbl, dtp_w, dtp_b, dtb);
  // 6) selective scan + Dskip + silu(z) gate -> bf16
  scan_kernel<<<12, 256, 0, stream>>>(dtb, xdbl, xc, xz, A_log, Dskip, yactb);
  // 7) x2 = x + yact @ out_proj^T   (K=768)
  gemm_bf16_wmma<false, true><<<dim3(Dm / 128, ROWS / 128), 256, 0, stream>>>(
      yactb, outpb, x2, nullptr, x, ROWS, Dm, DIN);
  // 8) x3 = LN2(x2) -> bf16
  ln_kernel_bf16<<<ROWS, 128, 0, stream>>>(x2, ln2_g, ln2_b, x3b);
  // 9) P = x3 @ W1a^T ; Q = x3 @ Wd^T   (both 8192x384, K=384)
  gemm_bf16_wmma<false, false><<<dim3(HID / 128, ROWS / 128), 256, 0, stream>>>(
      x3b, w1ab, Pm, nullptr, nullptr, ROWS, HID, Dm);
  gemm_bf16_wmma<false, false><<<dim3(HID / 128, ROWS / 128), 256, 0, stream>>>(
      x3b, wdb, Qm, nullptr, nullptr, ROWS, HID, Dm);
  // 10) gather + gelu + max over K -> bf16
  lcffn_gather_kernel<<<ROWS, 128, 0, stream>>>(Pm, Qm, fc1_b, idx, umb);
  // 11) out = x2 + um @ fc2^T + fc2_b
  gemm_bf16_wmma<true, true><<<dim3(Dm / 128, ROWS / 128), 256, 0, stream>>>(
      umb, fc2b, out, fc2_b, x2, ROWS, Dm, HID);
}
